// Attention_39453569581292
// MI455X (gfx1250) — matrix operand order repaired and checked by reading, not compiled, not run on hardware
//
#include <hip/hip_runtime.h>
#include <hip/hip_bf16.h>

// ---------------------------------------------------------------------------
// MI455X / gfx1250 (CDNA5, wave32) implementation.
// All large contractions run on V_WMMA_F32_16X16X32_F16 (f16 in, f32 accum).
// GEMM staging uses GLOBAL_LOAD_ASYNC_TO_LDS_B128 + double-buffered LDS when
// the toolchain exposes the builtin (ASYNCcnt / s_wait_asynccnt path).
// ---------------------------------------------------------------------------

typedef _Float16 half_t;
typedef __attribute__((ext_vector_type(16))) _Float16 v16h;
typedef __attribute__((ext_vector_type(8)))  _Float16 v8h;
typedef __attribute__((ext_vector_type(4)))  _Float16 v4h;
typedef __attribute__((ext_vector_type(8)))  float    v8f;
typedef __attribute__((__vector_size__(4 * sizeof(int)))) int v4i_t;

union FragH {
    v16h v;
    v8h  h2[2];
};

// ---- async global -> LDS (CDNA5, ASYNCcnt) --------------------------------
#if defined(__has_builtin)
#  if __has_builtin(__builtin_amdgcn_global_load_async_to_lds_b128)
#    define HAVE_ASYNC_LDS 1
#  endif
#endif
#ifndef HAVE_ASYNC_LDS
#  define HAVE_ASYNC_LDS 0
#endif

#if HAVE_ASYNC_LDS
__device__ __forceinline__ void async_g2l_b128(const half_t* g, half_t* l) {
    __builtin_amdgcn_global_load_async_to_lds_b128(
        (__attribute__((address_space(1))) v4i_t*)g,
        (__attribute__((address_space(3))) v4i_t*)l, 0, 0);
}
__device__ __forceinline__ void wait_async0() {
#if __has_builtin(__builtin_amdgcn_s_wait_asynccnt)
    __builtin_amdgcn_s_wait_asynccnt(0);
#else
    asm volatile("s_wait_asynccnt 0" ::: "memory");
#endif
}
#else
__device__ __forceinline__ void async_g2l_b128(const half_t* g, half_t* l) {
    *(v8h*)l = *(const v8h*)g;          // sync fallback: global load + ds store
}
__device__ __forceinline__ void wait_async0() {}
#endif

// ---- WMMA fragment loaders (CDNA5 ISA 7.12.2 layouts, wave32) -------------
// A (16x32 f16, row-major [M,K], leading dim ld in elements):
//   lanes 0-15:  M=lane,    K chunks [0..8) and [16..24)
//   lanes 16-31: M=lane-16, K chunks [8..16) and [24..32)
__device__ __forceinline__ FragH frag_a(const half_t* p, int ld, int lane) {
    int m  = lane & 15;
    int kg = (lane >> 4) << 3;          // 0 or 8
    FragH f;
    f.h2[0] = *(const v8h*)(p + m * ld + kg);
    f.h2[1] = *(const v8h*)(p + m * ld + kg + 16);
    return f;
}
// B (32x16 f16) consumed from memory laid out [N,K] (column n contiguous K):
//   lanes 0-15:  N=lane,    K chunks [0..8) and [16..24)
//   lanes 16-31: N=lane-16, K chunks [8..16) and [24..32)
__device__ __forceinline__ FragH frag_b(const half_t* p, int ld, int lane) {
    int n  = lane & 15;
    int kg = (lane >> 4) << 3;          // 0 or 8
    FragH f;
    f.h2[0] = *(const v8h*)(p + n * ld + kg);
    f.h2[1] = *(const v8h*)(p + n * ld + kg + 16);
    return f;
}

__device__ __forceinline__ v8f wmma16(FragH a, FragH b, v8f c) {
    return __builtin_amdgcn_wmma_f32_16x16x32_f16(
        /*neg_a=*/false, a.v, /*neg_b=*/false, b.v,
        /*c_mod=*/(short)0, c, /*reuse_a=*/false, /*reuse_b=*/false);
}

// ---- block-wide reductions (8 waves of 32) --------------------------------
__device__ __forceinline__ float blk_max(float v) {
    __shared__ float sm[8];
    #pragma unroll
    for (int off = 16; off > 0; off >>= 1) v = fmaxf(v, __shfl_xor(v, off, 32));
    if ((threadIdx.x & 31) == 0) sm[threadIdx.x >> 5] = v;
    __syncthreads();
    float m = sm[0];
    #pragma unroll
    for (int w = 1; w < 8; ++w) m = fmaxf(m, sm[w]);
    __syncthreads();
    return m;
}
__device__ __forceinline__ float blk_sum(float v) {
    __shared__ float sm[8];
    #pragma unroll
    for (int off = 16; off > 0; off >>= 1) v += __shfl_xor(v, off, 32);
    if ((threadIdx.x & 31) == 0) sm[threadIdx.x >> 5] = v;
    __syncthreads();
    float s = sm[0];
    #pragma unroll
    for (int w = 1; w < 8; ++w) s += sm[w];
    __syncthreads();
    return s;
}

// ---------------------------------------------------------------------------
// f32 -> f16 cast (vectorized x4)
// ---------------------------------------------------------------------------
__global__ void cast_f32_f16(const float* __restrict__ s, half_t* __restrict__ d, long n4) {
    long i = (long)blockIdx.x * blockDim.x + threadIdx.x;
    if (i < n4) {
        float4 v = ((const float4*)s)[i];
        v4h o;
        o[0] = (half_t)v.x; o[1] = (half_t)v.y; o[2] = (half_t)v.z; o[3] = (half_t)v.w;
        ((v4h*)d)[i] = o;
    }
}

// ---------------------------------------------------------------------------
// Generic WMMA GEMM: C[M,N] = A[M,K] * op(B) (+bias) (+activation)
//   B_NT = true : B stored [N,K]  (weights [out,in]); async-staged
//   B_NT = false: B stored [K,N]  (att@Wh, P@v); transpose-staged (sync)
// Block: 256 threads (8 waves), tile 128x128, K step 32. Waves in 4x2 grid,
// each wave owns a 32x64 sub-tile = 2x4 WMMA tiles (8 v_wmma per K step).
// Double-buffered LDS, one barrier per K step, async prefetch of tile i+1
// overlapped with WMMA on tile i. Batched via blockIdx.z = outer*nInner+inner.
// ---------------------------------------------------------------------------
enum { EPI_F32 = 0, EPI_F32_BIAS = 1, EPI_H = 2, EPI_H_BIAS = 3, EPI_H_ELU = 4 };

template <bool B_NT, int EPI>
__global__ __launch_bounds__(256) void gemm_f16(
    const half_t* __restrict__ A, const half_t* __restrict__ B, void* __restrict__ Cv,
    const float* __restrict__ bias,
    int M, int N, int K, int lda, int ldb, int ldc,
    long sAo, long sAi, long sBo, long sBi, long sCo, long sCi, int nInner)
{
    __shared__ half_t As[2][128 * 40];   // padded stride 40 halves = 80B (16B mult)
    __shared__ half_t Bs[2][128 * 40];

    const int tid  = threadIdx.x;
    const int lane = tid & 31;
    const int warp = tid >> 5;
    const int wm   = warp >> 1;          // 0..3  (rows, 32 each)
    const int wn   = warp & 1;           // 0..1  (cols, 64 each)
    const int tileM = blockIdx.y * 128;
    const int tileN = blockIdx.x * 128;

    const long zi = blockIdx.z % nInner;
    const long zo = blockIdx.z / nInner;
    A += zo * sAo + zi * sAi;
    B += zo * sBo + zi * sBi;
    half_t* Ch = (half_t*)Cv + zo * sCo + zi * sCi;
    float*  Cf = (float*)Cv  + zo * sCo + zi * sCi;

    // ---- staging helpers ---------------------------------------------------
    auto stageA = [&](int k0, int buf) {
        #pragma unroll
        for (int rep = 0; rep < 2; ++rep) {
            int idx = tid + rep * 256;            // 0..511
            int row = idx >> 2;                   // 0..127
            int kc  = (idx & 3) * 8;
            async_g2l_b128(A + (long)(tileM + row) * lda + k0 + kc,
                           &As[buf][row * 40 + kc]);
        }
    };
    auto stageB = [&](int k0, int buf) {
        if (B_NT) {
            int n   = tid >> 1;                   // 0..127
            int kc  = (tid & 1) * 16;
            int col = tileN + n;
            if (col < N) {
                async_g2l_b128(B + (long)col * ldb + k0 + kc,     &Bs[buf][n * 40 + kc]);
                async_g2l_b128(B + (long)col * ldb + k0 + kc + 8, &Bs[buf][n * 40 + kc + 8]);
            } else {
                v8h z = {};
                *(v8h*)&Bs[buf][n * 40 + kc]     = z;
                *(v8h*)&Bs[buf][n * 40 + kc + 8] = z;
            }
        } else {
            // B is [K,N]: coalesced global rows, transpose into LDS [N][K]
            #pragma unroll
            for (int rep = 0; rep < 2; ++rep) {
                int k   = (tid >> 4) + rep * 16;  // 0..31
                int n8  = (tid & 15) * 8;         // 0..120
                int col = tileN + n8;
                v8h bv = {};
                if (col + 8 <= N) {
                    bv = *(const v8h*)(B + (long)(k0 + k) * ldb + col);
                } else {
                    #pragma unroll
                    for (int i = 0; i < 8; ++i)
                        bv[i] = (col + i < N) ? B[(long)(k0 + k) * ldb + col + i]
                                              : (half_t)0.0f;
                }
                #pragma unroll
                for (int i = 0; i < 8; ++i)
                    Bs[buf][(n8 + i) * 40 + k] = bv[i];
            }
        }
    };

    v8f acc[2][4];
    #pragma unroll
    for (int i = 0; i < 2; ++i)
        #pragma unroll
        for (int j = 0; j < 4; ++j)
            acc[i][j] = (v8f){0.f, 0.f, 0.f, 0.f, 0.f, 0.f, 0.f, 0.f};

    const int nk = K >> 5;
    stageA(0, 0);
    stageB(0, 0);

    for (int i = 0; i < nk; ++i) {
        wait_async0();                    // tile i resident
        __syncthreads();                  // all waves done with buffer (i+1)&1
        if (i + 1 < nk) {                 // prefetch tile i+1 into other buffer
            stageA((i + 1) << 5, (i + 1) & 1);
            stageB((i + 1) << 5, (i + 1) & 1);
        }
        const half_t* Ab = As[i & 1];
        const half_t* Bb = Bs[i & 1];
        FragH a0 = frag_a(Ab + (wm * 32 +  0) * 40, 40, lane);
        FragH a1 = frag_a(Ab + (wm * 32 + 16) * 40, 40, lane);
        #pragma unroll
        for (int nt = 0; nt < 4; ++nt) {
            FragH bf = frag_b(Bb + (wn * 64 + nt * 16) * 40, 40, lane);
            acc[0][nt] = wmma16(a0, bf, acc[0][nt]);
            acc[1][nt] = wmma16(a1, bf, acc[1][nt]);
        }
    }

    // ---- epilogue (C/D layout: vgpr r -> M = r + 8*(lane>>4), N = lane&15) --
    const int nlo  = lane & 15;
    const int moff = (lane >> 4) * 8;
    #pragma unroll
    for (int mt = 0; mt < 2; ++mt) {
        #pragma unroll
        for (int nt = 0; nt < 4; ++nt) {
            int col = tileN + wn * 64 + nt * 16 + nlo;
            if (col >= N) continue;
            int rbase = tileM + wm * 32 + mt * 16 + moff;
            #pragma unroll
            for (int r = 0; r < 8; ++r) {
                float v = acc[mt][nt][r];
                if (EPI == EPI_F32_BIAS || EPI == EPI_H_BIAS) v += bias[col];
                if (EPI == EPI_H_ELU) v = v > 0.f ? v : __expf(v) - 1.f;
                long idx = (long)(rbase + r) * ldc + col;
                if (EPI == EPI_F32 || EPI == EPI_F32_BIAS) Cf[idx] = v;
                else                                       Ch[idx] = (half_t)v;
            }
        }
    }
}

// ---------------------------------------------------------------------------
// GAT e1/e2: e[l,b,n] = dot(Wh[b,n, l*1024 : +1024], a[l]) + a_b[l]
// grid (1024, L*B=24), block 256
// ---------------------------------------------------------------------------
__global__ __launch_bounds__(256) void gat_e_dots(
    const half_t* __restrict__ Wh, const float* __restrict__ ai, const float* __restrict__ aib,
    const float* __restrict__ aj, const float* __restrict__ ajb,
    float* __restrict__ e1, float* __restrict__ e2)
{
    int n = blockIdx.x, lb = blockIdx.y;
    int l = lb >> 3, b = lb & 7;
    const half_t* w  = Wh + ((long)(b * 1024 + n)) * 3072 + l * 1024;
    const float*  a1 = ai + l * 1024;
    const float*  a2 = aj + l * 1024;
    float s1 = 0.f, s2 = 0.f;
    for (int h = threadIdx.x; h < 1024; h += 256) {
        float wv = (float)w[h];
        s1 += wv * a1[h];
        s2 += wv * a2[h];
    }
    s1 = blk_sum(s1);
    s2 = blk_sum(s2);
    if (threadIdx.x == 0) {
        e1[lb * 1024 + n] = s1 + aib[l];
        e2[lb * 1024 + n] = s2 + ajb[l];
    }
}

// ---------------------------------------------------------------------------
// GAT attention rows: att[l,b,i,:] = softmax_j(adj[b,i,j]*lrelu(e1+e2[j]))
// grid (1024, 24), block 256, 4 cols/thread
// ---------------------------------------------------------------------------
__global__ __launch_bounds__(256) void gat_att_rows(
    const float* __restrict__ adj, const float* __restrict__ e1,
    const float* __restrict__ e2, half_t* __restrict__ att)
{
    int i = blockIdx.x, lb = blockIdx.y, b = lb & 7;
    const float* arow = adj + ((long)b * 1024 + i) * 1024;
    const float* e2r  = e2 + lb * 1024;
    float ei = e1[lb * 1024 + i];
    float v[4];
    float mx = -3.4e38f;
    #pragma unroll
    for (int q = 0; q < 4; ++q) {
        int j = threadIdx.x + q * 256;
        float x = ei + e2r[j];
        x = x > 0.f ? x : 0.2f * x;          // leaky_relu(ALPHA=0.2)
        x *= arow[j];
        v[q] = x;
        mx = fmaxf(mx, x);
    }
    mx = blk_max(mx);
    float s = 0.f;
    #pragma unroll
    for (int q = 0; q < 4; ++q) { v[q] = __expf(v[q] - mx); s += v[q]; }
    s = blk_sum(s);
    float inv = 1.f / s;
    half_t* orow = att + ((long)lb * 1024 + i) * 1024;
    #pragma unroll
    for (int q = 0; q < 4; ++q) orow[threadIdx.x + q * 256] = (half_t)(v[q] * inv);
}

// ---------------------------------------------------------------------------
// out-attention dots: eo[b,n] = dot(Who[b,n,:512], a) + a_b  (grid (1024,8))
// ---------------------------------------------------------------------------
__global__ __launch_bounds__(256) void out_e_dots(
    const half_t* __restrict__ Who, const float* __restrict__ ai, const float* __restrict__ aib,
    const float* __restrict__ aj, const float* __restrict__ ajb,
    float* __restrict__ eo1, float* __restrict__ eo2)
{
    int n = blockIdx.x, b = blockIdx.y;
    const half_t* w = Who + ((long)(b * 1024 + n)) * 512;
    float s1 = 0.f, s2 = 0.f;
    for (int h = threadIdx.x; h < 512; h += 256) {
        float wv = (float)w[h];
        s1 += wv * ai[h];
        s2 += wv * aj[h];
    }
    s1 = blk_sum(s1);
    s2 = blk_sum(s2);
    if (threadIdx.x == 0) {
        eo1[b * 1024 + n] = s1 + aib[0];
        eo2[b * 1024 + n] = s2 + ajb[0];
    }
}

// ---------------------------------------------------------------------------
// mask rows: a=softmax(adj*lrelu(eo1+eo2)); g=elu(a); mask=softmax(g) (f32 out)
// grid (1024, 8), block 256
// ---------------------------------------------------------------------------
__global__ __launch_bounds__(256) void mask_rows(
    const float* __restrict__ adj, const float* __restrict__ eo1,
    const float* __restrict__ eo2, float* __restrict__ mask)
{
    int i = blockIdx.x, b = blockIdx.y;
    const float* arow = adj + ((long)b * 1024 + i) * 1024;
    const float* e2r  = eo2 + b * 1024;
    float ei = eo1[b * 1024 + i];
    float v[4];
    float mx = -3.4e38f;
    #pragma unroll
    for (int q = 0; q < 4; ++q) {
        int j = threadIdx.x + q * 256;
        float x = ei + e2r[j];
        x = x > 0.f ? x : 0.2f * x;
        x *= arow[j];
        v[q] = x;
        mx = fmaxf(mx, x);
    }
    mx = blk_max(mx);
    float s = 0.f;
    #pragma unroll
    for (int q = 0; q < 4; ++q) { v[q] = __expf(v[q] - mx); s += v[q]; }
    s = blk_sum(s);
    float inv = 1.f / s;
    float g[4];
    float mx2 = -3.4e38f;
    #pragma unroll
    for (int q = 0; q < 4; ++q) {
        float a = v[q] * inv;
        float gg = a > 0.f ? a : __expf(a) - 1.f;   // elu
        g[q] = gg;
        mx2 = fmaxf(mx2, gg);
    }
    mx2 = blk_max(mx2);
    float s2 = 0.f;
    #pragma unroll
    for (int q = 0; q < 4; ++q) { g[q] = __expf(g[q] - mx2); s2 += g[q]; }
    s2 = blk_sum(s2);
    float inv2 = 1.f / s2;
    float* orow = mask + ((long)b * 1024 + i) * 1024;
    #pragma unroll
    for (int q = 0; q < 4; ++q) orow[threadIdx.x + q * 256] = g[q] * inv2;
}

// ---------------------------------------------------------------------------
// Fused attention scores: P = softmax_j(mask[b,i,j] * SCALE * (q . k)) -> f16
// One block = (b, h, 16 rows) x all 1024 cols. 8 waves, wave w owns cols
// [w*128, w*128+128): 8 WMMA col-tiles x 2 K-steps (HEAD_DIM=64), fragments
// loaded straight from global (q,k rows are [*,64] = natural [N,K]).
// Row softmax: shfl_xor over lanes sharing a row, then cross-wave via LDS.
// ---------------------------------------------------------------------------
__global__ __launch_bounds__(256) void attn_score_softmax(
    const half_t* __restrict__ qkv, const float* __restrict__ mask, half_t* __restrict__ P)
{
    __shared__ float red[8][16];
    const int rt = blockIdx.x, h = blockIdx.y, b = blockIdx.z;
    const int lane = threadIdx.x & 31, warp = threadIdx.x >> 5;
    const int nlo = lane & 15, hi = lane >> 4;
    const int LD = 1536;

    const half_t* q  = qkv + ((long)b * 1024 + rt * 16) * LD + h * 64;
    const half_t* kp = qkv + (long)b * 1024 * LD + 512 + h * 64;

    FragH a0 = frag_a(q, LD, lane);        // K 0..31
    FragH a1 = frag_a(q + 32, LD, lane);   // K 32..63

    v8f acc[8];
    #pragma unroll
    for (int ct = 0; ct < 8; ++ct) {
        const half_t* kb = kp + (long)(warp * 128 + ct * 16) * LD;
        FragH b0 = frag_b(kb, LD, lane);
        FragH b1 = frag_b(kb + 32, LD, lane);
        v8f c = (v8f){0.f, 0.f, 0.f, 0.f, 0.f, 0.f, 0.f, 0.f};
        c = wmma16(a0, b0, c);
        c = wmma16(a1, b1, c);
        acc[ct] = c;
    }

    const float* mrow = mask + ((long)b * 1024 + rt * 16) * 1024;
    float rmax[8];
    #pragma unroll
    for (int r = 0; r < 8; ++r) rmax[r] = -3.4e38f;
    #pragma unroll
    for (int ct = 0; ct < 8; ++ct) {
        #pragma unroll
        for (int r = 0; r < 8; ++r) {
            int row = r + 8 * hi;
            int col = warp * 128 + ct * 16 + nlo;
            float s = acc[ct][r] * 0.125f * mrow[(long)row * 1024 + col];
            acc[ct][r] = s;
            rmax[r] = fmaxf(rmax[r], s);
        }
    }
    #pragma unroll
    for (int off = 1; off < 16; off <<= 1) {
        #pragma unroll
        for (int r = 0; r < 8; ++r)
            rmax[r] = fmaxf(rmax[r], __shfl_xor(rmax[r], off, 32));
    }
    if (nlo == 0) {
        #pragma unroll
        for (int r = 0; r < 8; ++r) red[warp][r + 8 * hi] = rmax[r];
    }
    __syncthreads();
    float rowmax[8];
    #pragma unroll
    for (int r = 0; r < 8; ++r) {
        float m = red[0][r + 8 * hi];
        #pragma unroll
        for (int w = 1; w < 8; ++w) m = fmaxf(m, red[w][r + 8 * hi]);
        rowmax[r] = m;
    }
    __syncthreads();

    float rsum[8];
    #pragma unroll
    for (int r = 0; r < 8; ++r) rsum[r] = 0.f;
    #pragma unroll
    for (int ct = 0; ct < 8; ++ct) {
        #pragma unroll
        for (int r = 0; r < 8; ++r) {
            float p = __expf(acc[ct][r] - rowmax[r]);
            acc[ct][r] = p;
            rsum[r] += p;
        }
    }
    #pragma unroll
    for (int off = 1; off < 16; off <<= 1) {
        #pragma unroll
        for (int r = 0; r < 8; ++r) rsum[r] += __shfl_xor(rsum[r], off, 32);
    }
    if (nlo == 0) {
        #pragma unroll
        for (int r = 0; r < 8; ++r) red[warp][r + 8 * hi] = rsum[r];
    }
    __syncthreads();
    float rinv[8];
    #pragma unroll
    for (int r = 0; r < 8; ++r) {
        float s = red[0][r + 8 * hi];
        #pragma unroll
        for (int w = 1; w < 8; ++w) s += red[w][r + 8 * hi];
        rinv[r] = 1.f / s;
    }

    half_t* Pb = P + (((long)(b * 8 + h)) * 1024 + rt * 16) * 1024;
    #pragma unroll
    for (int ct = 0; ct < 8; ++ct) {
        #pragma unroll
        for (int r = 0; r < 8; ++r) {
            int col = warp * 128 + ct * 16 + nlo;
            Pb[(long)(r + 8 * hi) * 1024 + col] = (half_t)(acc[ct][r] * rinv[r]);
        }
    }
}

// ---------------------------------------------------------------------------
// Workspace layout (bytes)
// ---------------------------------------------------------------------------
static constexpr long NE_X    = 8L * 1024 * 512;
static constexpr long NE_QKVW = 1536L * 512;
static constexpr long NE_GATW = 3072L * 512;
static constexpr long NE_OUTW = 512L * 3072;
static constexpr long NE_PRJW = 512L * 512;
static constexpr long NE_QKV  = 8192L * 1536;
static constexpr long NE_WH   = 8192L * 3072;
static constexpr long NE_E    = 3L * 8 * 1024;
static constexpr long NE_ATT  = 24L * 1024 * 1024;
static constexpr long NE_HCAT = 8192L * 3072;
static constexpr long NE_WHO  = 8192L * 512;
static constexpr long NE_EO   = 8L * 1024;
static constexpr long NE_MASK = 8L * 1024 * 1024;
static constexpr long NE_P    = 64L * 1024 * 1024;
static constexpr long NE_AO   = 8192L * 512;

static constexpr size_t OFF_XH    = 0;
static constexpr size_t OFF_QKVWH = OFF_XH    + (size_t)NE_X    * 2;
static constexpr size_t OFF_GATWH = OFF_QKVWH + (size_t)NE_QKVW * 2;
static constexpr size_t OFF_OUTWH = OFF_GATWH + (size_t)NE_GATW * 2;
static constexpr size_t OFF_PRJWH = OFF_OUTWH + (size_t)NE_OUTW * 2;
static constexpr size_t OFF_QKVH  = OFF_PRJWH + (size_t)NE_PRJW * 2;
static constexpr size_t OFF_WHH   = OFF_QKVH  + (size_t)NE_QKV  * 2;
static constexpr size_t OFF_E1    = OFF_WHH   + (size_t)NE_WH   * 2;
static constexpr size_t OFF_E2    = OFF_E1    + (size_t)NE_E    * 4;
static constexpr size_t OFF_ATTH  = OFF_E2    + (size_t)NE_E    * 4;
static constexpr size_t OFF_HCATH = OFF_ATTH  + (size_t)NE_ATT  * 2;
static constexpr size_t OFF_WHOH  = OFF_HCATH + (size_t)NE_HCAT * 2;
static constexpr size_t OFF_EO1   = OFF_WHOH  + (size_t)NE_WHO  * 2;
static constexpr size_t OFF_EO2   = OFF_EO1   + (size_t)NE_EO   * 4;
static constexpr size_t OFF_MASK  = OFF_EO2   + (size_t)NE_EO   * 4;
static constexpr size_t OFF_P     = OFF_MASK  + (size_t)NE_MASK * 4;
static constexpr size_t OFF_AO    = OFF_P     + (size_t)NE_P    * 2;

extern "C" void kernel_launch(void* const* d_in, const int* in_sizes, int n_in,
                              void* d_out, int out_size, void* d_ws, size_t ws_size,
                              hipStream_t stream) {
    (void)in_sizes; (void)n_in; (void)out_size; (void)ws_size;

    const float* x      = (const float*)d_in[0];
    const float* adj    = (const float*)d_in[1];
    const float* qkv_w  = (const float*)d_in[2];
    const float* proj_w = (const float*)d_in[3];
    const float* proj_b = (const float*)d_in[4];
    const float* gat_W  = (const float*)d_in[5];
    const float* gat_Wb = (const float*)d_in[6];
    const float* gat_ai = (const float*)d_in[7];
    const float* gat_aib= (const float*)d_in[8];
    const float* gat_aj = (const float*)d_in[9];
    const float* gat_ajb= (const float*)d_in[10];
    const float* out_W  = (const float*)d_in[11];
    const float* out_Wb = (const float*)d_in[12];
    const float* out_ai = (const float*)d_in[13];
    const float* out_aib= (const float*)d_in[14];
    const float* out_aj = (const float*)d_in[15];
    const float* out_ajb= (const float*)d_in[16];

    char* ws = (char*)d_ws;
    half_t* x_h    = (half_t*)(ws + OFF_XH);
    half_t* qkvw_h = (half_t*)(ws + OFF_QKVWH);
    half_t* gatw_h = (half_t*)(ws + OFF_GATWH);
    half_t* outw_h = (half_t*)(ws + OFF_OUTWH);
    half_t* prjw_h = (half_t*)(ws + OFF_PRJWH);
    half_t* qkv_h  = (half_t*)(ws + OFF_QKVH);
    half_t* Wh_h   = (half_t*)(ws + OFF_WHH);
    float*  e1     = (float*) (ws + OFF_E1);
    float*  e2     = (float*) (ws + OFF_E2);
    half_t* att_h  = (half_t*)(ws + OFF_ATTH);
    half_t* hcat_h = (half_t*)(ws + OFF_HCATH);
    half_t* Who_h  = (half_t*)(ws + OFF_WHOH);
    float*  eo1    = (float*) (ws + OFF_EO1);
    float*  eo2    = (float*) (ws + OFF_EO2);
    float*  mask_f = (float*) (ws + OFF_MASK);
    half_t* P_h    = (half_t*)(ws + OFF_P);
    half_t* ao_h   = (half_t*)(ws + OFF_AO);

    auto cast = [&](const float* s, half_t* d, long n) {
        long n4 = n / 4;
        cast_f32_f16<<<dim3((unsigned)((n4 + 255) / 256)), dim3(256), 0, stream>>>(s, d, n4);
    };
    cast(x,      x_h,    NE_X);
    cast(qkv_w,  qkvw_h, NE_QKVW);
    cast(gat_W,  gatw_h, NE_GATW);
    cast(out_W,  outw_h, NE_OUTW);
    cast(proj_w, prjw_h, NE_PRJW);

    // G1: qkv = x @ qkv_w.T -> f16 [8192,1536]
    gemm_f16<true, EPI_H><<<dim3(12, 64, 1), 256, 0, stream>>>(
        x_h, qkvw_h, qkv_h, nullptr, 8192, 1536, 512, 512, 512, 1536,
        0, 0, 0, 0, 0, 0, 1);

    // G2: Wh = x @ gat_W.T + gat_Wb -> f16 [8192,3072] (col = l*1024 + h)
    gemm_f16<true, EPI_H_BIAS><<<dim3(24, 64, 1), 256, 0, stream>>>(
        x_h, gatw_h, Wh_h, gat_Wb, 8192, 3072, 512, 512, 512, 3072,
        0, 0, 0, 0, 0, 0, 1);

    // e1/e2 per (l,b,n)
    gat_e_dots<<<dim3(1024, 24), 256, 0, stream>>>(Wh_h, gat_ai, gat_aib, gat_aj, gat_ajb, e1, e2);

    // GAT attention rows -> att f16 [(l*8+b), i, j]
    gat_att_rows<<<dim3(1024, 24), 256, 0, stream>>>(adj, e1, e2, att_h);

    // G3: h_hat = elu(att @ Wh) per (l,b): z = l*8 + b (inner=b)
    gemm_f16<false, EPI_H_ELU><<<dim3(8, 8, 24), 256, 0, stream>>>(
        att_h, Wh_h, hcat_h, nullptr, 1024, 1024, 1024, 1024, 3072, 3072,
        /*sAo(l)*/ 8L * 1048576, /*sAi(b)*/ 1048576,
        /*sBo(l)*/ 1024,         /*sBi(b)*/ 1024L * 3072,
        /*sCo(l)*/ 1024,         /*sCi(b)*/ 1024L * 3072, 8);

    // G4: Who = h @ out_W.T + out_Wb -> f16 [8192,512]
    gemm_f16<true, EPI_H_BIAS><<<dim3(4, 64, 1), 256, 0, stream>>>(
        hcat_h, outw_h, Who_h, out_Wb, 8192, 512, 3072, 3072, 3072, 512,
        0, 0, 0, 0, 0, 0, 1);

    // eo1/eo2 per (b,n)
    out_e_dots<<<dim3(1024, 8), 256, 0, stream>>>(Who_h, out_ai, out_aib, out_aj, out_ajb, eo1, eo2);

    // mask = softmax(elu(softmax(adj*lrelu(eo1+eo2)))) -> f32 [8,1024,1024]
    mask_rows<<<dim3(1024, 8), 256, 0, stream>>>(adj, eo1, eo2, mask_f);

    // P = softmax(mask * scale * q k^T) per (b,h) -> f16 [64,1024,1024]
    attn_score_softmax<<<dim3(64, 8, 8), 256, 0, stream>>>(qkv_h, mask_f, P_h);

    // G6: out_attn = P @ v per (b,h): z = b*8 + h (inner=h), v at qkv col 1024
    gemm_f16<false, EPI_H><<<dim3(1, 8, 64), 256, 0, stream>>>(
        P_h, qkv_h + 1024, ao_h, nullptr, 1024, 64, 1024, 1024, 1536, 512,
        /*sAo(b)*/ 8L * 1048576, /*sAi(h)*/ 1048576,
        /*sBo(b)*/ 1024L * 1536, /*sBi(h)*/ 64,
        /*sCo(b)*/ 1024L * 512,  /*sCi(h)*/ 64, 8);

    // G7: out = out_attn @ proj_w.T + proj_b -> f32 d_out [8192,512]
    gemm_f16<true, EPI_F32_BIAS><<<dim3(4, 64, 1), 256, 0, stream>>>(
        ao_h, prjw_h, d_out, proj_b, 8192, 512, 512, 512, 512, 512,
        0, 0, 0, 0, 0, 0, 1);
}
